// VAERegressor_54417235640484
// MI455X (gfx1250) — matrix operand order repaired and checked by reading, not compiled, not run on hardware
//
#include <hip/hip_runtime.h>
#include <hip/hip_bf16.h>

// ---------------------------------------------------------------------------
// VGAE-GCN encoder (x2 graphs) + regression head for MI455X (gfx1250).
// Memory-bound: edge scatter dominates (~3.5 GB -> ~150us @ 23.3 TB/s).
// Dense GEMMs use v_wmma_f32_16x16x32_f16 with fragment-order LDS staging
// (fragments load as ds_load_b128 pairs, no scalar gathers). Edge scatter /
// elementwise passes are float4-vectorized (global_load_b128 +
// 4x global_atomic_add_f32 per quad) so the bandwidth-critical kernels stay
// issue-light; agg arrays (51MB) live in the 192MB L2 during the scatter.
// ---------------------------------------------------------------------------

typedef __attribute__((ext_vector_type(16))) _Float16 v16h;
typedef __attribute__((ext_vector_type(8)))  float    v8f;

__device__ __forceinline__ void atomAddF(float* p, float v) {
    __hip_atomic_fetch_add(p, v, __ATOMIC_RELAXED, __HIP_MEMORY_SCOPE_AGENT);
}

// ---------------------------------------------------------------- utilities
__global__ void zero_f32v(float4* __restrict__ p, long n4) {
    long i = (long)blockIdx.x * blockDim.x + threadIdx.x;
    if (i < n4) p[i] = make_float4(0.f, 0.f, 0.f, 0.f);
}

__global__ void deg_accum(const int* __restrict__ dst, float* __restrict__ deg, long E) {
    long i = (long)blockIdx.x * blockDim.x + threadIdx.x;
    if (i < E) atomAddF(&deg[dst[i]], 1.0f);
}

// in-place: buf holds deg, becomes dinv = rsqrt(deg + 1)  (self loop)
__global__ void finish_dinv(float* __restrict__ buf, int n) {
    int i = blockIdx.x * blockDim.x + threadIdx.x;
    if (i < n) buf[i] = rsqrtf(buf[i] + 1.0f);
}

// ------------------------------------------------------- generic WMMA GEMM
// C[M x N] = A[M x K] @ B[K x N]   (+bias, +relu optional), f32 in/out,
// f16 multiply with f32 accumulate. Workgroup tile: BM x BN, whole K staged
// (K <= KP, KP multiple of 32). (BM/16)*(BN/16) waves of 32 threads.
//
// LDS is laid out in WMMA *fragment order*: [tile][kchunk][lane][16 halves],
// so the inner loop is 2x ds_load_b128 per operand, straight into v_wmma.
// Wave32 fragment mapping (ISA 7.12.2):
//   A 16x32: lane = lhalf*16 + m;  e0..7 -> K (lhalf? 8..15 : 0..7),
//                                  e8..15 -> K 16 + (lhalf? 8..15 : 0..7)
//   B 32x16: lane = khalf*16 + n;  e0..15 -> K as for A, with khalf in place of lhalf
//   C/D 16x16: VGPR r -> row (r + lhalf*8), col = lane & 15
template<int BM, int BN, int KP, bool BIAS, bool RELU>
__global__ void gemm_wmma(const float* __restrict__ A, const float* __restrict__ B,
                          const float* __restrict__ bias, float* __restrict__ C,
                          int M, int K, int N) {
    constexpr int RT = BM / 16, CT = BN / 16, KT = KP / 32;
    constexpr int WAVES = RT * CT;
    constexpr int NTHREADS = WAVES * 32;
    static_assert(KP % 32 == 0, "KP must be multiple of 32");

    __shared__ alignas(32) _Float16 As[RT][KT][32][16];
    __shared__ alignas(32) _Float16 Bs[CT][KT][32][16];

    const int tid  = threadIdx.x;
    const int row0 = blockIdx.x * BM;
    const int col0 = blockIdx.y * BN;

    // stage A tile (f32 -> f16) into fragment order; zero-pad m>=M, k>=K
    for (int idx = tid; idx < BM * KP; idx += NTHREADS) {
        int r = idx / KP, k = idx % KP;
        int gr = row0 + r;
        float v = (gr < M && k < K) ? A[(long)gr * K + k] : 0.f;
        int rt = r >> 4, lm = r & 15;
        int kk = k >> 5, krem = k & 31;
        int lhalf = (krem >> 3) & 1;                  // K 8..15 / 24..31 -> high lanes
        int e     = (krem & 7) + ((krem >> 4) << 3);  // +8 for K >= 16 in chunk
        As[rt][kk][lhalf * 16 + lm][e] = (_Float16)v;
    }
    // stage B tile into fragment order
    for (int idx = tid; idx < KP * BN; idx += NTHREADS) {
        int k = idx / BN, c = idx % BN;
        float v = (k < K) ? B[(long)k * N + col0 + c] : 0.f;
        int ct = c >> 4, n = c & 15;
        int kk = k >> 5, krem = k & 31;
        int khalf = (krem >> 3) & 1, e = (krem & 7) + ((krem >> 4) << 3);
        Bs[ct][kk][khalf * 16 + n][e] = (_Float16)v;
    }
    __syncthreads();

    const int wave = tid >> 5;
    const int lane = tid & 31;
    const int rt_  = wave % RT;   // row tile of this wave
    const int ct_  = wave / RT;   // col tile of this wave

    v8f acc = {};
    #pragma unroll
    for (int kk = 0; kk < KT; ++kk) {
        v16h af = *(const v16h*)As[rt_][kk][lane];   // 2x ds_load_b128
        v16h bf = *(const v16h*)Bs[ct_][kk][lane];   // 2x ds_load_b128
        acc = __builtin_amdgcn_wmma_f32_16x16x32_f16(
            /*neg_a=*/false, af, /*neg_b=*/false, bf,
            /*c_mod=*/(short)0, acc, /*reuse_a=*/false, /*reuse_b=*/false);
    }

    // epilogue: lane-fixed column, 8 rows; one base pointer, stride-N steps
    const int lhalf = lane >> 4, lm = lane & 15;
    const int m0 = row0 + rt_ * 16 + lhalf * 8;
    const int n  = col0 + ct_ * 16 + lm;
    float bval = 0.f;
    if constexpr (BIAS) bval = bias[n];
    float* cp = C + (long)m0 * N + n;

    if (row0 + BM <= M) {                 // full tile: unguarded stores
        #pragma unroll
        for (int r = 0; r < 8; ++r) {
            float v = acc[r];
            if constexpr (BIAS) v += bval;
            if constexpr (RELU) v = fmaxf(v, 0.f);
            cp[(long)r * N] = v;
        }
    } else {                              // tail tile: guarded
        #pragma unroll
        for (int r = 0; r < 8; ++r) {
            if (m0 + r < M) {
                float v = acc[r];
                if constexpr (BIAS) v += bval;
                if constexpr (RELU) v = fmaxf(v, 0.f);
                cp[(long)r * N] = v;
            }
        }
    }
}

// ------------------------------------------------ edge scatter-accumulate
// agg[dst, f] += h[src, f] * dinv[src] * dinv[dst]
// One thread per (edge, feature-quad): b128 gather + 4 consecutive f32
// atomics off one base address; indices/dinv loaded once per quad (4x fewer
// redundant loads than per-feature threading).
template<int F>
__global__ void edge_agg(const int* __restrict__ src, const int* __restrict__ dst,
                         const float* __restrict__ dinv, const float* __restrict__ h,
                         float* __restrict__ agg, long E) {
    constexpr int Q = F / 4;
    long idx = (long)blockIdx.x * blockDim.x + threadIdx.x;
    if (idx >= E * (long)Q) return;
    long e = idx / Q;
    int  q = (int)(idx - e * Q);
    int s = src[e], d = dst[e];
    float nrm = dinv[s] * dinv[d];
    const float4 hv = *(const float4*)(h + (long)s * F + q * 4);  // global_load_b128
    float* ap = agg + (long)d * F + q * 4;
    atomAddF(ap + 0, hv.x * nrm);
    atomAddF(ap + 1, hv.y * nrm);
    atomAddF(ap + 2, hv.z * nrm);
    atomAddF(ap + 3, hv.w * nrm);
}

// out = [relu](agg + h * dinv_i^2 + b)   (self-loop term + bias, float4-wide)
template<int F, bool RELU>
__global__ void combine(const float* __restrict__ agg, const float* __restrict__ h0,
                        const float* __restrict__ dinv, const float* __restrict__ b,
                        float* __restrict__ out, int n) {
    constexpr int Q = F / 4;
    long idx = (long)blockIdx.x * blockDim.x + threadIdx.x;
    if (idx >= (long)n * Q) return;
    long i = idx / Q;
    int  q = (int)(idx - i * Q);
    float di = dinv[i];
    float d2 = di * di;
    long base = i * F + q * 4;
    float4 a  = *(const float4*)(agg + base);
    float4 hv = *(const float4*)(h0 + base);
    float4 bv = *(const float4*)(b + q * 4);
    float4 o;
    o.x = a.x + hv.x * d2 + bv.x;
    o.y = a.y + hv.y * d2 + bv.y;
    o.z = a.z + hv.z * d2 + bv.z;
    o.w = a.w + hv.w * d2 + bv.w;
    if (RELU) {
        o.x = fmaxf(o.x, 0.f); o.y = fmaxf(o.y, 0.f);
        o.z = fmaxf(o.z, 0.f); o.w = fmaxf(o.w, 0.f);
    }
    *(float4*)(out + base) = o;
}

// ------------------------------------------------------------ mean pooling
// z is [G][64]; side A writes cols 0..31, side B cols 32..63 (LAT = 32).
// One thread per (node, quad): b128 load + 4 atomics.
__global__ void pool_accum(const float* __restrict__ mu, const int* __restrict__ batch,
                           float* __restrict__ z, float* __restrict__ cnt,
                           int colofs, int n) {
    long idx = (long)blockIdx.x * blockDim.x + threadIdx.x;
    if (idx >= (long)n * 8) return;          // 32 feats = 8 quads
    long i = idx >> 3;
    int  q = (int)(idx & 7);
    int  g = batch[i];
    float4 v = *(const float4*)(mu + i * 32 + q * 4);
    float* zp = z + (long)g * 64 + colofs + q * 4;
    atomAddF(zp + 0, v.x);
    atomAddF(zp + 1, v.y);
    atomAddF(zp + 2, v.z);
    atomAddF(zp + 3, v.w);
    if (q == 0) atomAddF(&cnt[g], 1.0f);
}

__global__ void pool_div(float* __restrict__ z, const float* __restrict__ cnt,
                         int colofs, int G) {
    int idx = blockIdx.x * blockDim.x + threadIdx.x;
    if (idx >= G * 32) return;
    int g = idx >> 5, f = idx & 31;
    z[(long)g * 64 + colofs + f] /= fmaxf(cnt[g], 1.0f);
}

// --------------------------------------------------------- final 32->2 layer
__global__ void head3(const float* __restrict__ h2, const float* __restrict__ W,
                      const float* __restrict__ b, float* __restrict__ out, int G) {
    int idx = blockIdx.x * blockDim.x + threadIdx.x;
    if (idx >= G * 2) return;
    int g = idx >> 1, o = idx & 1;
    float s = b[o];
    #pragma unroll
    for (int k = 0; k < 32; ++k) s += h2[(long)g * 32 + k] * W[k * 2 + o];
    out[idx] = s;
}

// ---------------------------------------------------------------------------
static inline int gridFor(long n) { return (int)((n + 255) / 256); }

// zero helper: n must be a multiple of 4 (true for all buffers here)
static inline void zeroN(float* p, long n, hipStream_t stream) {
    long n4 = n / 4;
    zero_f32v<<<gridFor(n4), 256, 0, stream>>>((float4*)p, n4);
}

extern "C" void kernel_launch(void* const* d_in, const int* in_sizes, int n_in,
                              void* d_out, int out_size, void* d_ws, size_t ws_size,
                              hipStream_t stream) {
    const float* xA   = (const float*)d_in[0];
    const int*   eiA  = (const int*)  d_in[1];
    const int*   batA = (const int*)  d_in[2];
    const float* xB   = (const float*)d_in[3];
    const int*   eiB  = (const int*)  d_in[4];
    const int*   batB = (const int*)  d_in[5];
    const float* W1   = (const float*)d_in[6];
    const float* b1   = (const float*)d_in[7];
    const float* Wmu  = (const float*)d_in[8];
    const float* bmu  = (const float*)d_in[9];
    const float* Wr1  = (const float*)d_in[10];
    const float* br1  = (const float*)d_in[11];
    const float* Wr2  = (const float*)d_in[12];
    const float* br2  = (const float*)d_in[13];
    const float* Wr3  = (const float*)d_in[14];
    const float* br3  = (const float*)d_in[15];

    const int  N  = in_sizes[2];          // 100000 nodes
    const long E  = in_sizes[1] / 2;      // 800000 edges
    const int  G  = out_size / 2;         // 2048 graphs
    const int  F0 = in_sizes[0] / N;      // 58 node features
    const int  HID = 128, LAT = 32;

    // ---- workspace partition (floats), ~129 MB total
    float* H0   = (float*)d_ws;                   // [N*128] gemm1 out
    float* B1   = H0   + (size_t)N * HID;         // [N*128] agg1 / h1 (in place)
    float* M0   = B1   + (size_t)N * HID;         // [N*32]  gemm2 out
    float* B2   = M0   + (size_t)N * LAT;         // [N*32]  agg2 / mu (in place)
    float* dinv = B2   + (size_t)N * LAT;         // [N] deg -> dinv in place
    float* z    = dinv + (size_t)N;               // [G*64] pooled (A|B)
    float* cntA = z    + (size_t)G * 64;          // [G]
    float* cntB = cntA + (size_t)G;               // [G]
    float* hh1  = cntB + (size_t)G;               // [G*64]
    float* hh2  = hh1  + (size_t)G * 64;          // [G*32]
    (void)ws_size; (void)n_in;

    // zero accumulation targets shared across both sides (fresh every call)
    zeroN(z,    (long)G * 64, stream);
    zeroN(cntA, G,            stream);
    zeroN(cntB, G,            stream);

    auto run_side = [&](const float* x, const int* ei, const int* bat,
                        float* cnt, int colofs) {
        const int* src = ei;       // edge_index row 0
        const int* dst = ei + E;   // edge_index row 1

        // degree -> dinv
        zeroN(dinv, N, stream);
        deg_accum<<<gridFor(E), 256, 0, stream>>>(dst, dinv, E);
        finish_dinv<<<gridFor(N), 256, 0, stream>>>(dinv, N);

        // conv1: H0 = X @ W1     (K=58 pad 64, N=128), one 16x128 stripe / WG
        gemm_wmma<16, 128, 64, false, false>
            <<<dim3((N + 15) / 16, 1), 256, 0, stream>>>(x, W1, nullptr, H0, N, F0, HID);
        zeroN(B1, (long)N * HID, stream);
        edge_agg<128><<<gridFor(E * 32), 256, 0, stream>>>(src, dst, dinv, H0, B1, E);
        combine<128, true><<<gridFor((long)N * 32), 256, 0, stream>>>(B1, H0, dinv, b1, B1, N);

        // conv2: M0 = H1 @ Wmu   (K=128, N=32), 64x32 tile / WG
        gemm_wmma<64, 32, 128, false, false>
            <<<dim3((N + 63) / 64, 1), 256, 0, stream>>>(B1, Wmu, nullptr, M0, N, HID, LAT);
        zeroN(B2, (long)N * LAT, stream);
        edge_agg<32><<<gridFor(E * 8), 256, 0, stream>>>(src, dst, dinv, M0, B2, E);
        combine<32, false><<<gridFor((long)N * 8), 256, 0, stream>>>(B2, M0, dinv, bmu, B2, N);

        // global mean pool into z[:, colofs:colofs+32]
        pool_accum<<<gridFor((long)N * 8), 256, 0, stream>>>(B2, bat, z, cnt, colofs, N);
        pool_div<<<gridFor((long)G * 32), 256, 0, stream>>>(z, cnt, colofs, G);
    };

    run_side(xA, eiA, batA, cntA, 0);
    run_side(xB, eiB, batB, cntB, 32);

    // regression head: relu(z@Wr1+br1) -> relu(@Wr2+br2) -> @Wr3+br3
    gemm_wmma<32, 64, 64, true, true>
        <<<dim3((G + 31) / 32, 1), 256, 0, stream>>>(z, Wr1, br1, hh1, G, 64, 64);
    gemm_wmma<64, 32, 64, true, true>
        <<<dim3((G + 63) / 64, 1), 256, 0, stream>>>(hh1, Wr2, br2, hh2, G, 64, 32);
    head3<<<gridFor((long)G * 2), 256, 0, stream>>>(hh2, Wr3, br3, (float*)d_out, G);
}
